// CustomLanguageModel_21706764714472
// MI455X (gfx1250) — matrix operand order repaired and checked by reading, not compiled, not run on hardware
//
#include <hip/hip_runtime.h>
#include <hip/hip_bf16.h>

typedef __attribute__((ext_vector_type(16))) _Float16 v16h;
typedef __attribute__((ext_vector_type(8)))  float    v8f;
typedef unsigned int u32x4 __attribute__((ext_vector_type(4)));
typedef int          i32x4 __attribute__((ext_vector_type(4)));
typedef int          i32x8 __attribute__((ext_vector_type(8)));

#define E_  384
#define T_  256
#define H_  6
#define DH_ 64
#define V_  65
#define L_  6

// ---------------- wave32 reductions ----------------
__device__ inline float wave_reduce_sum(float v) {
#pragma unroll
  for (int m = 16; m > 0; m >>= 1) v += __shfl_xor(v, m, 32);
  return v;
}
__device__ inline float wave_reduce_max(float v) {
#pragma unroll
  for (int m = 16; m > 0; m >>= 1) v = fmaxf(v, __shfl_xor(v, m, 32));
  return v;
}

// ---------------- LDS raw offset (strip generic-aperture base) ----------------
typedef __attribute__((address_space(3))) const char* lds_cptr_t;
__device__ inline unsigned lds_off(const void* p) {
  return (unsigned)(unsigned long long)(lds_cptr_t)p;
}

// ---------------- Tensor Data Mover: 2D f16 tile -> LDS ----------------
// D# per CDNA5 ISA ch.8: group0 = {count/lds_addr/global_addr/type}, group1 =
// {data_size, tensor dims (for OOB zero-fill), tile dims, row stride}.
// Toolchain exposes the 6-arg builtin: (g0, g1, g2, g3, extra, cpol).
__device__ inline void tdm_load_tile_f16(unsigned lds_addr, const _Float16* g,
                                         unsigned rem_w, unsigned rem_h,
                                         unsigned long long row_stride_elems,
                                         unsigned tile_w, unsigned tile_h)
{
  unsigned long long ga = (unsigned long long)g;
  u32x4 g0;
  g0[0] = 1u;                                   // count=1, user descriptor
  g0[1] = lds_addr;                             // LDS byte address
  g0[2] = (unsigned)ga;                         // global_addr[31:0]
  g0[3] = (unsigned)((ga >> 32) & 0x1FFFFFFull) // global_addr[56:32]
          | 0x80000000u;                        // type=2 ("image")
  i32x8 g1;
  g1[0] = (int)(1u << 16);                                   // data_size=1 -> 2 bytes
  g1[1] = (int)((rem_w & 0xFFFFu) << 16);                    // tensor_dim0[15:0]
  g1[2] = (int)((rem_w >> 16) | ((rem_h & 0xFFFFu) << 16));  // dim0[31:16] | dim1[15:0]
  g1[3] = (int)((rem_h >> 16) | (tile_w << 16));             // dim1[31:16] | tile_dim0
  g1[4] = (int)(tile_h & 0xFFFFu);                           // tile_dim1 (tile_dim2=0)
  g1[5] = (int)(unsigned)(row_stride_elems & 0xFFFFFFFFull); // dim0_stride[31:0]
  g1[6] = (int)(unsigned)((row_stride_elems >> 32) & 0xFFFFull); // dim0_stride[47:32]
  g1[7] = 0;                                                 // dim1_stride (unused, 2D)
  i32x4 zz4 = {0, 0, 0, 0};
  i32x8 zz8 = {0, 0, 0, 0, 0, 0, 0, 0};
  __builtin_amdgcn_tensor_load_to_lds(g0, g1, zz4, zz4, zz8, 0);
}

// ---------------- generic WMMA GEMM with TDM-staged, double-buffered tiles ----
// C = act(A @ Wt^T + bias) (+resid).  A: [M,K] f16 row-major (lda==K).
// Wt: [N,K] f16 row-major (pre-transposed weights), + z*wz batch offset.
// Block = 64x64 C tile, 8 waves, each wave 16x32. K-loop step 32, 2 LDS buffers.
template <bool OUTH>
__global__ __launch_bounds__(256) void gemm_wmma(
    const _Float16* __restrict__ A, int lda,
    const _Float16* __restrict__ Wt, long wz,
    const float* __restrict__ bias,
    const float* __restrict__ resid,
    float* __restrict__ Cf, _Float16* __restrict__ Ch,
    int ldc, int czcol,
    int M, int N, int K, int relu)
{
  __shared__ _Float16 Ash[2][64 * 32];
  __shared__ _Float16 Wsh[2][64 * 32];

  int z = blockIdx.z;
  const _Float16* Wz = Wt + (long)z * wz;
  int colz = z * czcol;
  int row0 = blockIdx.y * 64;
  int col0 = blockIdx.x * 64;
  int tid = threadIdx.x, wid = tid >> 5, lane = tid & 31;
  int mrow = (wid & 3) * 16;   // wave's rows in [0,64)
  int ncol = (wid >> 2) * 32;  // wave's cols in [0,64)
  int m = lane & 15, hi = lane >> 4;

  const int nk = K / 32;
  if (wid == 0) {  // prologue: stage first K-slab via TDM (issued once per block)
    tdm_load_tile_f16(lds_off(&Ash[0][0]), A + (long)row0 * lda,
                      (unsigned)K, (unsigned)(M - row0), (unsigned long long)lda, 32, 64);
    tdm_load_tile_f16(lds_off(&Wsh[0][0]), Wz + (long)col0 * K,
                      (unsigned)K, (unsigned)(N - col0), (unsigned long long)K, 32, 64);
  }

  v8f acc0 = {};
  v8f acc1 = {};

  for (int step = 0; step < nk; ++step) {
    int cur = step & 1;
    if (wid == 0) __builtin_amdgcn_s_wait_tensorcnt((short)0);
    __syncthreads();  // staged slab visible to all waves
    if (wid == 0 && step + 1 < nk) {  // prefetch next slab into other buffer
      int kk = (step + 1) * 32;
      tdm_load_tile_f16(lds_off(&Ash[cur ^ 1][0]), A + (long)row0 * lda + kk,
                        (unsigned)(K - kk), (unsigned)(M - row0),
                        (unsigned long long)lda, 32, 64);
      tdm_load_tile_f16(lds_off(&Wsh[cur ^ 1][0]), Wz + (long)col0 * K + kk,
                        (unsigned)(K - kk), (unsigned)(N - col0),
                        (unsigned long long)K, 32, 64);
    }
    // A fragment (16x32): one contiguous 32B run per lane
    const _Float16* arow = &Ash[cur][(mrow + m) * 32];
    v16h a;
#pragma unroll
    for (int r = 0; r < 8; ++r) {
      int kx = hi * 16 + 2 * r;
      a[2 * r]     = arow[kx];
      a[2 * r + 1] = arow[kx + 1];
    }
    // B fragments (32x16): K contiguous per n-row in transposed tile (32B run)
    const _Float16* b0p = &Wsh[cur][(ncol + m) * 32 + hi * 16];
    const _Float16* b1p = &Wsh[cur][(ncol + 16 + m) * 32 + hi * 16];
    v16h b0, b1;
#pragma unroll
    for (int i = 0; i < 16; ++i) { b0[i] = b0p[i]; b1[i] = b1p[i]; }

    acc0 = __builtin_amdgcn_wmma_f32_16x16x32_f16(false, a, false, b0, (short)0, acc0, false, false);
    acc1 = __builtin_amdgcn_wmma_f32_16x16x32_f16(false, a, false, b1, (short)0, acc1, false, false);
    // no tail barrier needed: next overwrite of 'cur' happens only after the
    // next iteration's top barrier
  }

  // epilogue: C/D layout -> m = r + hi*8, n = lane&15
#pragma unroll
  for (int f = 0; f < 2; ++f) {
    v8f& acc = f ? acc1 : acc0;
    int col = col0 + ncol + f * 16 + (lane & 15);
    if (col < N) {
      float bv = bias ? bias[col] : 0.f;
#pragma unroll
      for (int r = 0; r < 8; ++r) {
        int row = row0 + mrow + r + hi * 8;
        if (row < M) {
          float vv = acc[r] + bv;
          if (relu) vv = fmaxf(vv, 0.f);
          long ci = (long)row * ldc + colz + col;
          if (resid) vv += resid[ci];
          if constexpr (OUTH) Ch[ci] = (_Float16)vv;
          else                Cf[ci] = vv;
        }
      }
    }
  }
}

// ---------------- attention: full score matrix in 258KB LDS, WMMA both matmuls
// one block per (b,h); q/k/v f16 [B,T,H,DH]; att output f16 [B,T,H*DH]
__global__ __launch_bounds__(256) void attn_flash(
    const _Float16* __restrict__ q, const _Float16* __restrict__ k,
    const _Float16* __restrict__ v, _Float16* __restrict__ att)
{
  constexpr int SST = T_ + 2;
  extern __shared__ float S[];
  int bh = blockIdx.x;
  int b = bh / H_, h = bh % H_;
  long base = ((long)b * T_ * H_ + h) * DH_;
  const _Float16* qb = q + base;
  const _Float16* kb = k + base;
  const _Float16* vb = v + base;
  int tid = threadIdx.x, wid = tid >> 5, lane = tid & 31;
  int m = lane & 15, hi = lane >> 4;
  const float scale = 0.125f;  // DH^-0.5

  // phase 1: S[t,s] = scale * q.k^T (causal tiles only)
  for (int ti = wid; ti < 16; ti += 8) {
    for (int si = 0; si <= ti; ++si) {
      v8f acc = {};
#pragma unroll
      for (int kk = 0; kk < DH_; kk += 32) {
        const _Float16* qrow = qb + (ti * 16 + m) * (H_ * DH_) + kk;
        v16h a;
#pragma unroll
        for (int r = 0; r < 8; ++r) {
          int kx = hi * 16 + 2 * r;
          a[2 * r]     = qrow[kx];
          a[2 * r + 1] = qrow[kx + 1];
        }
        const _Float16* krow = kb + (si * 16 + m) * (H_ * DH_) + kk;  // n = s
        v16h bf;
#pragma unroll
        for (int r = 0; r < 8; ++r) {
          int kx = hi * 16 + 2 * r;
          bf[2 * r]     = krow[kx];
          bf[2 * r + 1] = krow[kx + 1];
        }
        acc = __builtin_amdgcn_wmma_f32_16x16x32_f16(false, a, false, bf, (short)0, acc, false, false);
      }
#pragma unroll
      for (int r = 0; r < 8; ++r) {
        int t = ti * 16 + r + hi * 8;
        int s = si * 16 + m;
        float sv = acc[r] * scale;
        if (s > t) sv = -1e30f;
        S[t * SST + s] = sv;
      }
    }
  }
  __syncthreads();

  // phase 2: row softmax in LDS; zero-pad to 32-boundary for P@V
  for (int t = wid * 32; t < wid * 32 + 32; ++t) {
    float* row = S + t * SST;
    float mx = -1e30f;
    for (int s = lane; s <= t; s += 32) mx = fmaxf(mx, row[s]);
    mx = wave_reduce_max(mx);
    float sum = 0.f;
    for (int s = lane; s <= t; s += 32) {
      float e = __expf(row[s] - mx);
      row[s] = e;
      sum += e;
    }
    sum = wave_reduce_sum(sum);
    float inv = 1.f / sum;
    for (int s = lane; s <= t; s += 32) row[s] *= inv;
    int end = ((t >> 5) + 1) << 5;
    for (int s = t + 1 + lane; s < end; s += 32) row[s] = 0.f;
  }
  __syncthreads();

  // phase 3: att = P @ V
  for (int ti = wid; ti < 16; ti += 8) {
    v8f a0 = {}, a1 = {}, a2 = {}, a3 = {};
    for (int sp = 0; sp <= ti / 2; ++sp) {
      const float* prow = S + (ti * 16 + m) * SST + sp * 32;
      v16h a;
#pragma unroll
      for (int r = 0; r < 8; ++r) {
        int kx = hi * 16 + 2 * r;
        a[2 * r]     = (_Float16)prow[kx];
        a[2 * r + 1] = (_Float16)prow[kx + 1];
      }
#pragma unroll
      for (int dt = 0; dt < 4; ++dt) {
        const _Float16* vcol = vb + (long)(sp * 32) * (H_ * DH_) + dt * 16 + m;  // n = d
        v16h bf;
#pragma unroll
        for (int r = 0; r < 8; ++r) {
          int kx = hi * 16 + 2 * r;
          bf[2 * r]     = vcol[(long)kx * (H_ * DH_)];
          bf[2 * r + 1] = vcol[(long)(kx + 1) * (H_ * DH_)];
        }
        v8f& acc = dt == 0 ? a0 : dt == 1 ? a1 : dt == 2 ? a2 : a3;
        acc = __builtin_amdgcn_wmma_f32_16x16x32_f16(false, a, false, bf, (short)0, acc, false, false);
      }
    }
#pragma unroll
    for (int dt = 0; dt < 4; ++dt) {
      v8f& acc = dt == 0 ? a0 : dt == 1 ? a1 : dt == 2 ? a2 : a3;
#pragma unroll
      for (int r = 0; r < 8; ++r) {
        int t = ti * 16 + r + hi * 8;
        att[((long)(b * T_ + t)) * E_ + h * DH_ + dt * 16 + m] = (_Float16)acc[r];
      }
    }
  }
}

// ---------------- layernorm: one wave per row of E=384, f16 output ------------
__global__ __launch_bounds__(256) void layernorm_k(
    const float* __restrict__ x, const float* __restrict__ g,
    const float* __restrict__ bta, _Float16* __restrict__ out, int rows)
{
  int row = blockIdx.x * 8 + threadIdx.y;
  if (row >= rows) return;
  int lane = threadIdx.x;
  const float* xr = x + (long)row * E_;
  float vals[12];
  float s = 0.f;
#pragma unroll
  for (int j = 0; j < 12; ++j) { vals[j] = xr[lane + 32 * j]; s += vals[j]; }
  s = wave_reduce_sum(s);
  float mean = s * (1.f / E_);
  float vs = 0.f;
#pragma unroll
  for (int j = 0; j < 12; ++j) { float d = vals[j] - mean; vs += d * d; }
  vs = wave_reduce_sum(vs);
  float rstd = rsqrtf(vs * (1.f / E_) + 1e-5f);
  _Float16* orow = out + (long)row * E_;
#pragma unroll
  for (int j = 0; j < 12; ++j) {
    int e = lane + 32 * j;
    orow[e] = (_Float16)((vals[j] - mean) * rstd * g[e] + bta[e]);
  }
}

// ---------------- weight transpose + f16 convert: dst[N,K] = src[K,N] ---------
__global__ void transpose_f16(const float* __restrict__ src, _Float16* __restrict__ dst,
                              int K, int N)
{
  long mat = (long)blockIdx.z * K * N;
  int idx = blockIdx.x * 256 + threadIdx.x;
  if (idx < K * N) {
    int n = idx / K, k = idx - n * K;
    dst[mat + idx] = (_Float16)src[mat + (long)k * N + n];
  }
}

// ---------------- embedding ----------------
__global__ void embed_k(const int* __restrict__ idx, const float* __restrict__ tok,
                        const float* __restrict__ pos, float* __restrict__ x, int BT)
{
  int bt = blockIdx.x;
  if (bt >= BT) return;
  int t = bt % T_;
  int id = idx[bt];
  const float* te = tok + (long)id * E_;
  const float* pe = pos + (long)t * E_;
  float* xr = x + (long)bt * E_;
  for (int e = threadIdx.x; e < E_; e += blockDim.x) xr[e] = te[e] + pe[e];
}

// ---------------- loss: deterministic two-stage reduction ----------------
__global__ void loss_part1(const float* __restrict__ logits, const int* __restrict__ tgt,
                           float* __restrict__ partials)
{
  __shared__ float ps[8];
  int row = blockIdx.x * 8 + threadIdx.y;
  int lane = threadIdx.x;
  const float* lr = logits + (long)row * V_;
  float mx = -1e30f;
#pragma unroll
  for (int j = 0; j < 3; ++j) { int e = lane + 32 * j; if (e < V_) mx = fmaxf(mx, lr[e]); }
  mx = wave_reduce_max(mx);
  float sum = 0.f;
#pragma unroll
  for (int j = 0; j < 3; ++j) { int e = lane + 32 * j; if (e < V_) sum += __expf(lr[e] - mx); }
  sum = wave_reduce_sum(sum);
  if (lane == 0) {
    float lse = mx + __logf(sum);
    ps[threadIdx.y] = lse - lr[tgt[row]];
  }
  __syncthreads();
  if (threadIdx.y == 0 && lane == 0) {
    float s = 0.f;
#pragma unroll
    for (int i = 0; i < 8; ++i) s += ps[i];
    partials[blockIdx.x] = s;
  }
}

__global__ void loss_part2(const float* __restrict__ partials, float* __restrict__ out,
                           int n, float invBT)
{
  __shared__ float red[256];
  int tid = threadIdx.x;
  float s = 0.f;
  for (int i = tid; i < n; i += 256) s += partials[i];
  red[tid] = s;
  __syncthreads();
  for (int off = 128; off > 0; off >>= 1) {
    if (tid < off) red[tid] += red[tid + off];
    __syncthreads();
  }
  if (tid == 0) out[0] = red[0] * invBT;
}

// ---------------- host driver ----------------
extern "C" void kernel_launch(void* const* d_in, const int* in_sizes, int n_in,
                              void* d_out, int out_size, void* d_ws, size_t ws_size,
                              hipStream_t stream)
{
  (void)n_in; (void)out_size; (void)ws_size;
  const int*   idx  = (const int*)d_in[0];
  const int*   tgt  = (const int*)d_in[1];
  const float* tok  = (const float*)d_in[2];
  const float* pos  = (const float*)d_in[3];
  const float* Wq   = (const float*)d_in[4];
  const float* Wk   = (const float*)d_in[5];
  const float* Wv   = (const float*)d_in[6];
  const float* Wo   = (const float*)d_in[7];
  const float* bo   = (const float*)d_in[8];
  const float* ln1s = (const float*)d_in[9];
  const float* ln1b = (const float*)d_in[10];
  const float* ln2s = (const float*)d_in[11];
  const float* ln2b = (const float*)d_in[12];
  const float* W1   = (const float*)d_in[13];
  const float* b1   = (const float*)d_in[14];
  const float* W2   = (const float*)d_in[15];
  const float* b2   = (const float*)d_in[16];
  const float* lnfs = (const float*)d_in[17];
  const float* lnfb = (const float*)d_in[18];
  const float* Wout = (const float*)d_in[19];
  const float* bout = (const float*)d_in[20];

  int BT = in_sizes[0];  // B*T
  int B  = BT / T_;

  char* wp = (char*)d_ws;
  size_t szF  = (size_t)BT * E_ * sizeof(float);
  size_t szH  = (size_t)BT * E_ * sizeof(_Float16);
  size_t szFF = (size_t)BT * 4 * E_ * sizeof(_Float16);
  float*    x     = (float*)wp;    wp += szF;
  _Float16* h16   = (_Float16*)wp; wp += szH;
  _Float16* att16 = (_Float16*)wp; wp += szH;
  _Float16* q16   = (_Float16*)wp; wp += szH;
  _Float16* k16   = (_Float16*)wp; wp += szH;
  _Float16* v16   = (_Float16*)wp; wp += szH;
  _Float16* ff16  = (_Float16*)wp; wp += szFF;
  // pre-transposed f16 weights
  size_t szQKVw = (size_t)L_ * H_ * E_ * DH_ * sizeof(_Float16);
  size_t szWo   = (size_t)L_ * E_ * E_ * sizeof(_Float16);
  size_t szW1   = (size_t)L_ * E_ * 4 * E_ * sizeof(_Float16);
  _Float16* wtq   = (_Float16*)wp; wp += szQKVw;
  _Float16* wtk   = (_Float16*)wp; wp += szQKVw;
  _Float16* wtv   = (_Float16*)wp; wp += szQKVw;
  _Float16* wto   = (_Float16*)wp; wp += szWo;
  _Float16* wt1   = (_Float16*)wp; wp += szW1;
  _Float16* wt2   = (_Float16*)wp; wp += szW1;
  _Float16* wtout = (_Float16*)wp; wp += (size_t)E_ * V_ * sizeof(_Float16);
  float* partials = (float*)wp;

  const int attn_lds = T_ * (T_ + 2) * (int)sizeof(float);  // 264192 B < 320KB WGP LDS
  (void)hipFuncSetAttribute((const void*)attn_flash,
                            hipFuncAttributeMaxDynamicSharedMemorySize, attn_lds);

  // ---- pre-transpose weights to f16 [N,K] ----
  {
    int eh = E_ * DH_;  // per-head [E,DH] -> [DH,E]
    transpose_f16<<<dim3((eh + 255) / 256, 1, L_ * H_), 256, 0, stream>>>(Wq, wtq, E_, DH_);
    transpose_f16<<<dim3((eh + 255) / 256, 1, L_ * H_), 256, 0, stream>>>(Wk, wtk, E_, DH_);
    transpose_f16<<<dim3((eh + 255) / 256, 1, L_ * H_), 256, 0, stream>>>(Wv, wtv, E_, DH_);
    int ee = E_ * E_;
    transpose_f16<<<dim3((ee + 255) / 256, 1, L_), 256, 0, stream>>>(Wo, wto, E_, E_);
    int e4 = E_ * 4 * E_;
    transpose_f16<<<dim3((e4 + 255) / 256, 1, L_), 256, 0, stream>>>(W1, wt1, E_, 4 * E_);
    transpose_f16<<<dim3((e4 + 255) / 256, 1, L_), 256, 0, stream>>>(W2, wt2, 4 * E_, E_);
    int ev = E_ * V_;
    transpose_f16<<<dim3((ev + 255) / 256, 1, 1), 256, 0, stream>>>(Wout, wtout, E_, V_);
  }

  dim3 lnGrid(BT / 8), lnBlk(32, 8);

  embed_k<<<BT, 256, 0, stream>>>(idx, tok, pos, x, BT);

  for (int l = 0; l < L_; ++l) {
    layernorm_k<<<lnGrid, lnBlk, 0, stream>>>(x, ln1s + l * E_, ln1b + l * E_, h16, BT);

    // QKV: batched over heads via blockIdx.z; per-head Wt is [DH,E]
    dim3 gq(1, BT / 64, H_);
    long hb = (long)l * H_ * DH_ * E_;
    gemm_wmma<true><<<gq, 256, 0, stream>>>(h16, E_, wtq + hb, (long)DH_ * E_,
        nullptr, nullptr, nullptr, q16, E_, DH_, BT, DH_, E_, 0);
    gemm_wmma<true><<<gq, 256, 0, stream>>>(h16, E_, wtk + hb, (long)DH_ * E_,
        nullptr, nullptr, nullptr, k16, E_, DH_, BT, DH_, E_, 0);
    gemm_wmma<true><<<gq, 256, 0, stream>>>(h16, E_, wtv + hb, (long)DH_ * E_,
        nullptr, nullptr, nullptr, v16, E_, DH_, BT, DH_, E_, 0);

    attn_flash<<<B * H_, 256, attn_lds, stream>>>(q16, k16, v16, att16);

    // x = x + att @ Wo + bo
    gemm_wmma<false><<<dim3(E_ / 64, BT / 64, 1), 256, 0, stream>>>(
        att16, E_, wto + (long)l * E_ * E_, 0, bo + l * E_, x, x, nullptr,
        E_, 0, BT, E_, E_, 0);

    layernorm_k<<<lnGrid, lnBlk, 0, stream>>>(x, ln2s + l * E_, ln2b + l * E_, h16, BT);

    // ff16 = relu(h @ W1 + b1)
    gemm_wmma<true><<<dim3(4 * E_ / 64, BT / 64, 1), 256, 0, stream>>>(
        h16, E_, wt1 + (long)l * 4 * E_ * E_, 0, b1 + (long)l * 4 * E_,
        nullptr, nullptr, ff16, 4 * E_, 0, BT, 4 * E_, E_, 1);

    // x = x + ff16 @ W2 + b2
    gemm_wmma<false><<<dim3(E_ / 64, BT / 64, 1), 256, 0, stream>>>(
        ff16, 4 * E_, wt2 + (long)l * 4 * E_ * E_, 0, b2 + l * E_, x, x, nullptr,
        E_, 0, BT, E_, 4 * E_, 0);
  }

  layernorm_k<<<lnGrid, lnBlk, 0, stream>>>(x, lnfs, lnfb, h16, BT);

  // logits -> d_out (N=65: TDM zero-fills OOB rows, stores are column-guarded)
  float* logits = (float*)d_out;
  gemm_wmma<false><<<dim3((V_ + 63) / 64, BT / 64, 1), 256, 0, stream>>>(
      h16, E_, wtout, 0, bout, nullptr, logits, nullptr, V_, 0, BT, V_, E_, 0);

  loss_part1<<<dim3(BT / 8), dim3(32, 8), 0, stream>>>(logits, tgt, partials);
  loss_part2<<<1, 256, 0, stream>>>(partials, logits + (size_t)BT * V_, BT / 8, 1.0f / BT);
}
